// HybridMemory_26826365731010
// MI455X (gfx1250) — matrix operand order repaired and checked by reading, not compiled, not run on hardware
//
#include <hip/hip_runtime.h>
#include <hip/hip_bf16.h>

typedef __attribute__((ext_vector_type(16))) __bf16 v16bf;
typedef __attribute__((ext_vector_type(8)))  __bf16 v8bf;
typedef __attribute__((ext_vector_type(8)))  float  v8f;
typedef __attribute__((ext_vector_type(4)))  float  v4f;

#define B_   256
#define D_   2048
#define N_   65536
#define C_   8192
#define INV_TEMP 20.0f

#define TM 128   // M tile per block (B rows)
#define TN 128   // N tile per block (feature rows)
#define TK 64    // K chunk staged in LDS (2 wmma K=32 steps)

// ---------------------------------------------------------------- kernel 0
__global__ void k_init(float* __restrict__ simsum, float* __restrict__ diag,
                       float* __restrict__ nums, int* __restrict__ targets,
                       const int* __restrict__ indexes, const int* __restrict__ labels,
                       float* __restrict__ out) {
  int gid = blockIdx.x * blockDim.x + threadIdx.x;
  const int total = C_ * B_;
  for (int i = gid; i < total; i += gridDim.x * blockDim.x) simsum[i] = 0.0f;
  if (gid < C_) nums[gid] = 0.0f;
  if (gid < B_) { diag[gid] = 0.0f; targets[gid] = labels[indexes[gid]]; }
  if (gid == 0) out[0] = 0.0f;
}

// ---------------------------------------------------------------- kernel 1
__global__ void k_counts(float* __restrict__ nums, const int* __restrict__ labels) {
  int j = blockIdx.x * blockDim.x + threadIdx.x;
  if (j < N_) atomicAdd(&nums[labels[j]], 1.0f);
}

// ---------------------------------------------------------------- kernel 2
// sims tile = inputs[Mtile,K] @ features[Ntile,K]^T via v_wmma_f32_16x16x32_bf16.
// Epilogue fuses: segment-sum of sims/T into simsum[C,B], and diag[b] += sims^2/T
// for entries whose label matches targets[b] (algebraic replacement of the
// centers GEMM: diag = sum_j mask * sims^2 / T).
__launch_bounds__(256, 2)
__global__ void k_gemm(const float* __restrict__ inputs,
                       const float* __restrict__ features,
                       const int* __restrict__ labels,
                       const int* __restrict__ targets,
                       float* __restrict__ simsum,
                       float* __restrict__ diag) {
  __shared__ __bf16 As[TM * TK];
  __shared__ __bf16 Bs[TN * TK];

  const int tid  = threadIdx.x;
  const int lane = tid & 31;
  const int wave = tid >> 5;
  const int wm = wave & 3;          // 4 waves along M -> 32 rows each
  const int wn = wave >> 2;         // 2 waves along N -> 64 cols each
  const int msub = wm * 32;         // 2 wmma M tiles
  const int nsub = wn * 64;         // 4 wmma N tiles
  const int m_base = blockIdx.y * TM;
  const int n_base = blockIdx.x * TN;

  v8f acc[2][4];
#pragma unroll
  for (int mt = 0; mt < 2; ++mt)
#pragma unroll
    for (int nt = 0; nt < 4; ++nt) acc[mt][nt] = (v8f)0.0f;

  // staging: 256 threads, 128 rows x 64 cols per array -> each thread: one
  // half-row (32 floats) of A and of B per K chunk.
  const int srow = tid >> 1;
  const int scol = (tid & 1) * 32;
  const float* ga = inputs   + (size_t)(m_base + srow) * D_ + scol;
  const float* gb = features + (size_t)(n_base + srow) * D_ + scol;
  __bf16* sa = &As[srow * TK + scol];
  __bf16* sb = &Bs[srow * TK + scol];

  for (int k0 = 0; k0 < D_; k0 += TK) {
    const float* pa = ga + k0;
    const float* pb = gb + k0;
    if (k0 + TK < D_) {               // warm next chunk: global_prefetch_b8
      __builtin_prefetch(pa + TK, 0, 1);
      __builtin_prefetch(pb + TK, 0, 1);
    }
#pragma unroll
    for (int u = 0; u < 4; ++u) {     // 8 floats -> 8 bf16 per step, x4
      v4f a0 = *(const v4f*)(pa + u * 8);
      v4f a1 = *(const v4f*)(pa + u * 8 + 4);
      v4f b0 = *(const v4f*)(pb + u * 8);
      v4f b1 = *(const v4f*)(pb + u * 8 + 4);
      v8bf va, vb;
#pragma unroll
      for (int e = 0; e < 4; ++e) {
        va[e] = (__bf16)a0[e]; va[4 + e] = (__bf16)a1[e];
        vb[e] = (__bf16)b0[e]; vb[4 + e] = (__bf16)b1[e];
      }
      *(v8bf*)(sa + u * 8) = va;      // ds_store_b128
      *(v8bf*)(sb + u * 8) = vb;
    }
    __syncthreads();

#pragma unroll
    for (int ks = 0; ks < TK; ks += 32) {
      union Frag { v16bf v; v8bf h[2]; };
      Frag afr[2], bfr[4];
      // A 16x32 bf16 layout: lanes 0-15 hold K {0..7,16..23}, lanes 16-31
      // hold K {8..15,24..31}; row = lane%16.
      const int koff = (lane & 16) ? 8 : 0;
#pragma unroll
      for (int mt = 0; mt < 2; ++mt) {
        const __bf16* base = &As[(msub + mt * 16 + (lane & 15)) * TK + ks];
        afr[mt].h[0] = *(const v8bf*)(base + koff);
        afr[mt].h[1] = *(const v8bf*)(base + 16 + koff);
      }
      // B 32x16 bf16 layout: lanes 0-15 hold K {0..7,16..23}, lanes 16-31
      // K {8..15,24..31}; col = lane%16; contiguous K in memory (features row-major in K).
      const int kb = (lane & 16) ? 8 : 0;
#pragma unroll
      for (int nt = 0; nt < 4; ++nt) {
        const __bf16* base = &Bs[(nsub + nt * 16 + (lane & 15)) * TK + ks + kb];
        bfr[nt].h[0] = *(const v8bf*)(base);
        bfr[nt].h[1] = *(const v8bf*)(base + 16);
      }
#pragma unroll
      for (int mt = 0; mt < 2; ++mt)
#pragma unroll
        for (int nt = 0; nt < 4; ++nt)
          acc[mt][nt] = __builtin_amdgcn_wmma_f32_16x16x32_bf16(
              false, afr[mt].v, false, bfr[nt].v, (short)0, acc[mt][nt],
              false, false);
    }
    __syncthreads();
  }

  // ---- fused epilogue: scatter into segment-sum + masked diag ----
  // C/D layout: VGPR r, lane l -> M = r + (l>=16 ? 8 : 0), N = l%16.
  const int mrow0 = m_base + msub + ((lane & 16) ? 8 : 0);
  int tb[16];
#pragma unroll
  for (int mt = 0; mt < 2; ++mt)
#pragma unroll
    for (int r = 0; r < 8; ++r) tb[mt * 8 + r] = targets[mrow0 + mt * 16 + r];

#pragma unroll
  for (int nt = 0; nt < 4; ++nt) {
    const int j = n_base + nsub + nt * 16 + (lane & 15);
    const int c = labels[j];
    float* col = simsum + (size_t)c * B_;
#pragma unroll
    for (int mt = 0; mt < 2; ++mt) {
#pragma unroll
      for (int r = 0; r < 8; ++r) {
        const int b = mrow0 + mt * 16 + r;
        const float s = acc[mt][nt][r];       // raw similarity
        const float v = s * INV_TEMP;         // logits
        atomicAdd(&col[b], v);                // segment_sum(logits.T)[c][b]
        if (tb[mt * 8 + r] == c) atomicAdd(&diag[b], v * s);  // sims^2/T
      }
    }
  }
}

// ---------------------------------------------------------------- kernel 3
__global__ void k_loss(const float* __restrict__ simsum,
                       const float* __restrict__ diag,
                       const float* __restrict__ nums,
                       const int* __restrict__ targets,
                       float* __restrict__ out) {
  __shared__ float red[256];
  const int b = blockIdx.x;
  const int tgt = targets[b];
  const float dg = diag[b];
  float local = 0.0f;
  for (int c = threadIdx.x; c < C_; c += blockDim.x) {
    const float n = nums[c];
    if (n > 0.0f) {
      const float val = (c == tgt) ? dg : simsum[(size_t)c * B_ + b] / n;
      local += expf(val);                    // unstabilized, per reference
    }
  }
  red[threadIdx.x] = local;
  __syncthreads();
  for (int s = 128; s > 0; s >>= 1) {
    if (threadIdx.x < s) red[threadIdx.x] += red[threadIdx.x + s];
    __syncthreads();
  }
  if (threadIdx.x == 0) {
    const float denom = red[0] + 1e-6f;
    const float p = expf(dg) / denom + 1e-6f;
    atomicAdd(out, -logf(p) / (float)B_);
  }
}

// ---------------------------------------------------------------- launch
extern "C" void kernel_launch(void* const* d_in, const int* in_sizes, int n_in,
                              void* d_out, int out_size, void* d_ws, size_t ws_size,
                              hipStream_t stream) {
  const float* inputs   = (const float*)d_in[0];
  const int*   indexes  = (const int*)d_in[1];
  const float* features = (const float*)d_in[2];
  const int*   labels   = (const int*)d_in[3];
  float* out = (float*)d_out;

  char* ws = (char*)d_ws;
  float* simsum  = (float*)ws;                                  // C_*B_ floats (8 MB)
  float* diag    = (float*)(ws + sizeof(float) * (size_t)C_ * B_);  // B_ floats
  float* nums    = diag + B_;                                   // C_ floats
  int*   targets = (int*)(nums + C_);                           // B_ ints

  k_init<<<dim3(8192), dim3(256), 0, stream>>>(simsum, diag, nums, targets,
                                               indexes, labels, out);
  k_counts<<<dim3(N_ / 256), dim3(256), 0, stream>>>(nums, labels);
  k_gemm<<<dim3(N_ / TN, B_ / TM), dim3(256), 0, stream>>>(
      inputs, features, labels, targets, simsum, diag);
  k_loss<<<dim3(B_), dim3(256), 0, stream>>>(simsum, diag, nums, targets, out);
}
